// PostProcess_60232621359697
// MI455X (gfx1250) — compile-verified, matrix operand order checked by reading, not run on hardware
//
#include <hip/hip_runtime.h>
#include <hip/hip_bf16.h>

// ---------------------------------------------------------------------------
// SSD300/COCO post-processing for MI455X (gfx1250, wave32).
// WMMA (v_wmma_f32_16x16x32_f16) computes the softmax row-sums (16 priors /
// wave, ones-matrix reduction) with branchless clause-batched loads.
// Everything else is latency-bound scalar/VALU work: radix-select top-1000,
// bitonic sort, LDS-tiled IoU, bitmask greedy NMS.
// Workspace requirement: ~10.6 MB.
// ---------------------------------------------------------------------------

typedef __attribute__((ext_vector_type(16))) _Float16 v16h;
typedef __attribute__((ext_vector_type(8)))  float    v8f;

constexpr int B = 16;
constexpr int P = 8732;
constexpr int C = 81;          // classes incl background
constexpr int NC1 = 80;        // foreground classes
constexpr int KCAND = 1024;    // padded PRE_NMS_TOPK (reference uses 1000)
constexpr int KTOP = 1000;
constexpr int DETS = 100;
constexpr float SCORE_THRESH = 0.05f;
constexpr float NMS_THRESH = 0.45f;
constexpr float MIN_SIZE = 0.01f;
constexpr int HBINS = 65536;
constexpr int TILES_PER_IMG = (P + 15) / 16;   // 546

// ---------------------------------------------------------------------------
// Kernel 1: box decode (SSD coder, scaled to pixel coords)
// ---------------------------------------------------------------------------
__global__ void k_decode(const float* __restrict__ rel,
                         const float* __restrict__ priors,
                         const int* __restrict__ tsz,
                         float* __restrict__ boxes) {
    int t = blockIdx.x * blockDim.x + threadIdx.x;
    if (t >= B * P) return;
    int b = t / P, p = t % P;
    float h = (float)tsz[b * 2 + 0];
    float w = (float)tsz[b * 2 + 1];
    const float* r  = rel + (size_t)t * 4;
    const float* pr = priors + (size_t)p * 4;
    float cx = pr[0] + r[0] * 0.1f * pr[2];
    float cy = pr[1] + r[1] * 0.1f * pr[3];
    float bw = pr[2] * __expf(r[2] * 0.2f);
    float bh = pr[3] * __expf(r[3] * 0.2f);
    float* o = boxes + (size_t)t * 4;
    o[0] = (cx - 0.5f * bw) * w;
    o[1] = (cy - 0.5f * bh) * h;
    o[2] = (cx + 0.5f * bw) * w;
    o[3] = (cy + 0.5f * bh) * h;
}

// ---------------------------------------------------------------------------
// Kernel 2: softmax max + sum(exp) per prior via WMMA row-reduction.
// Each wave owns a 16-prior tile. A = exp(logits - max) as f16 (16x32 tiles,
// K padded 81 -> 96), B = all-ones 32x16 => D[m][*] = rowsum(m) in f32.
// ISA 16-bit A layout: lanes 0-15 row m=lane, K in {0..7,16..23}+32c;
// lanes 16-31 row m=lane-16, K in {8..15,24..31}+32c. Per lane that is six
// contiguous 8-float runs at K = 16*r + kbase, loaded branchlessly (clamped
// indices) so the 48 loads issue as one VMEM clause with a single wait.
// C/D layout: VGPR v, lanes 0-15 -> M=v; lanes 16-31 -> M=v+8.
// ---------------------------------------------------------------------------
__global__ void k_softmax_wmma(const float* __restrict__ logits,
                               float* __restrict__ sumexp,
                               float* __restrict__ maxlog) {
    int wave = (blockIdx.x * blockDim.x + threadIdx.x) >> 5;
    int lane = threadIdx.x & 31;
    if (wave >= B * TILES_PER_IMG) return;          // wave-uniform
    int b  = wave / TILES_PER_IMG;
    int p0 = (wave % TILES_PER_IMG) * 16;
    int row = lane & 15;
    int p = p0 + row;
    bool pvalid = (p < P);
    int pc = pvalid ? p : (P - 1);                  // clamped: loads always legal
    const float* lg = logits + ((size_t)b * P + pc) * C;
    int kbase = (lane >> 4) * 8;                    // 0 or 8

    // ---- unconditional clamped loads: 6 runs x 8 floats -> one clause ----
    float raw[48];
#pragma unroll
    for (int r = 0; r < 6; ++r) {
#pragma unroll
        for (int e = 0; e < 8; ++e) {
            int k = 16 * r + kbase + e;
            raw[r * 8 + e] = lg[(k < C) ? k : (C - 1)];
        }
    }

    // ---- branchless row max (invalid K -> -inf-ish) ----
    float mx = -3.0e38f;
#pragma unroll
    for (int r = 0; r < 6; ++r) {
#pragma unroll
        for (int e = 0; e < 8; ++e) {
            int k = 16 * r + kbase + e;
            float v = (k < C) ? raw[r * 8 + e] : -3.0e38f;
            mx = fmaxf(mx, v);
        }
    }
    mx = fmaxf(mx, __shfl_xor(mx, 16, 32));

    v16h ones;
#pragma unroll
    for (int e = 0; e < 16; ++e) ones[e] = (_Float16)1.0f;

    v8f acc = {};
#pragma unroll
    for (int c = 0; c < 3; ++c) {
        v16h a;
#pragma unroll
        for (int e = 0; e < 16; ++e) {
            // chunk c element e: run r = 2c + (e>>3), offset e&7
            int r  = 2 * c + (e >> 3);
            int ee = e & 7;
            int k  = 16 * r + kbase + ee;
            float t = __expf(raw[r * 8 + ee] - mx);
            float v = (pvalid && k < C) ? t : 0.0f;   // v_cndmask, no branch
            a[e] = (_Float16)v;
        }
        acc = __builtin_amdgcn_wmma_f32_16x16x32_f16(
            false, a, false, ones, (short)0, acc, false, false);
    }

    if (lane == 0 || lane == 16) {
        int rbase = (lane == 0) ? 0 : 8;
#pragma unroll
        for (int v = 0; v < 8; ++v) {
            int pp = p0 + rbase + v;
            if (pp < P) sumexp[(size_t)b * P + pp] = acc[v];
        }
    }
    if (lane < 16 && pvalid) maxlog[(size_t)b * P + p] = mx;
}

// ---------------------------------------------------------------------------
// Kernel 3: zero histogram + counters
// ---------------------------------------------------------------------------
__global__ void k_zero(unsigned* __restrict__ hist,
                       int* __restrict__ c1, int* __restrict__ c2) {
    int t = blockIdx.x * blockDim.x + threadIdx.x;
    if (t < B * HBINS) hist[t] = 0u;
    if (t < B) { c1[t] = 0; c2[t] = 0; }
}

// ---------------------------------------------------------------------------
// Kernel 4: per-image 65536-bin histogram of passing score bit-patterns.
// Positive floats compare identically to their uint bit patterns.
// ---------------------------------------------------------------------------
__global__ void k_hist(const float* __restrict__ logits,
                       const float* __restrict__ sumexp,
                       const float* __restrict__ maxlog,
                       const float* __restrict__ boxes,
                       unsigned* __restrict__ hist) {
    int t = blockIdx.x * blockDim.x + threadIdx.x;
    if (t >= B * P) return;
    int b = t / P;
    const float* lg = logits + (size_t)t * C;
    __builtin_prefetch(lg + C, 0, 1);
    const float* bx = boxes + (size_t)t * 4;
    if (!((bx[2] - bx[0] >= MIN_SIZE) && (bx[3] - bx[1] >= MIN_SIZE))) return;
    float mx = maxlog[t], inv = 1.0f / sumexp[t];
    unsigned* h = hist + (size_t)b * HBINS;
#pragma unroll 8
    for (int c = 1; c < C; ++c) {
        float s = __expf(lg[c] - mx) * inv;
        if (s > SCORE_THRESH) atomicAdd(&h[__float_as_uint(s) >> 16], 1u);
    }
}

// ---------------------------------------------------------------------------
// Kernel 5: find the bin t* at which the top-down cumulative count reaches
// KTOP. count(top16 > t*) < KTOP. t* == 0 means "take every valid key".
// ---------------------------------------------------------------------------
__global__ void k_findt(const unsigned* __restrict__ hist,
                        int* __restrict__ tstar) {
    __shared__ unsigned partial[1024];
    int b = blockIdx.x, tid = threadIdx.x;
    const unsigned* h = hist + (size_t)b * HBINS;
    unsigned s = 0;
    for (int i = 0; i < 64; ++i) s += h[tid * 64 + i];
    partial[tid] = s;
    __syncthreads();
    if (tid == 0) {
        unsigned cum = 0; int ts = 0;
        for (int seg = 1023; seg >= 0; --seg) {
            if (cum + partial[seg] >= (unsigned)KTOP) {
                for (int i = seg * 64 + 63; i >= seg * 64; --i) {
                    if (cum + h[i] >= (unsigned)KTOP) { ts = i; break; }
                    cum += h[i];
                }
                break;
            }
            cum += partial[seg];
        }
        tstar[b] = ts;
    }
}

// ---------------------------------------------------------------------------
// Kernel 6: collect candidates. list1: keys strictly above bin t* (count
// guaranteed < KTOP). list2: keys in bin t* (capacity-clamped).
// ---------------------------------------------------------------------------
__global__ void k_collect(const float* __restrict__ logits,
                          const float* __restrict__ sumexp,
                          const float* __restrict__ maxlog,
                          const float* __restrict__ boxes,
                          const int* __restrict__ tstar,
                          int* __restrict__ cnt1, int* __restrict__ cnt2,
                          float* __restrict__ l1v, int* __restrict__ l1i,
                          float* __restrict__ l2v, int* __restrict__ l2i) {
    int t = blockIdx.x * blockDim.x + threadIdx.x;
    if (t >= B * P) return;
    int b = t / P, p = t % P;
    const float* lg = logits + (size_t)t * C;
    const float* bx = boxes + (size_t)t * 4;
    if (!((bx[2] - bx[0] >= MIN_SIZE) && (bx[3] - bx[1] >= MIN_SIZE))) return;
    float mx = maxlog[t], inv = 1.0f / sumexp[t];
    int ts = tstar[b];
#pragma unroll 8
    for (int c = 1; c < C; ++c) {
        float s = __expf(lg[c] - mx) * inv;
        if (s > SCORE_THRESH) {
            int tt = (int)(__float_as_uint(s) >> 16);
            int idx = p * NC1 + (c - 1);
            if (tt > ts) {
                int pos = atomicAdd(&cnt1[b], 1);
                if (pos < KCAND) {
                    l1v[b * KCAND + pos] = s;
                    l1i[b * KCAND + pos] = idx;
                }
            } else if (ts > 0 && tt == ts) {
                int pos = atomicAdd(&cnt2[b], 1);
                if (pos < KCAND) {
                    l2v[b * KCAND + pos] = s;
                    l2i[b * KCAND + pos] = idx;
                }
            }
        }
    }
}

// ---------------------------------------------------------------------------
// Kernel 7: per-image 2048-wide bitonic sort by (score desc, idx asc) -- the
// same tie-break as lax.top_k. Then gather boxes/labels, reduce max_coord,
// and apply per-class coordinate offsets for batched NMS.
// ---------------------------------------------------------------------------
__global__ void k_sort(const int* __restrict__ cnt1, const int* __restrict__ cnt2,
                       const float* __restrict__ l1v, const int* __restrict__ l1i,
                       const float* __restrict__ l2v, const int* __restrict__ l2i,
                       const float* __restrict__ boxes,
                       float* __restrict__ sval, int* __restrict__ sidx,
                       float* __restrict__ cboxes, int* __restrict__ clabels,
                       float* __restrict__ oboxes) {
    __shared__ float sv[2048];
    __shared__ int   si[2048];
    __shared__ float red[1024];
    int b = blockIdx.x, tid = threadIdx.x;
    int n1 = min(cnt1[b], KCAND);
    int n2 = min(cnt2[b], KCAND);
    for (int i = tid; i < 2048; i += 1024) {
        float v = -1.0f; int id = 0x7FFFFFFF;
        if (i < n1)           { v = l1v[b * KCAND + i];        id = l1i[b * KCAND + i]; }
        else if (i < n1 + n2) { v = l2v[b * KCAND + (i - n1)]; id = l2i[b * KCAND + (i - n1)]; }
        sv[i] = v; si[i] = id;
    }
    __syncthreads();
    for (int k = 2; k <= 2048; k <<= 1) {
        for (int j = k >> 1; j > 0; j >>= 1) {
            for (int i = tid; i < 2048; i += 1024) {
                int ixj = i ^ j;
                if (ixj > i) {
                    float va = sv[i], vb = sv[ixj];
                    int   ia = si[i], ib = si[ixj];
                    bool desc = ((i & k) == 0);
                    bool swp = desc ? ((va < vb) || (va == vb && ia > ib))
                                    : ((va > vb) || (va == vb && ia < ib));
                    if (swp) { sv[i] = vb; sv[ixj] = va; si[i] = ib; si[ixj] = ia; }
                }
            }
            __syncthreads();
        }
    }
    // emit top KCAND (entries >= KTOP invalidated, matching PRE_NMS_TOPK=1000)
    int e = tid;
    float v = (e < KTOP) ? sv[e] : -1.0f;
    int id = si[e];
    sval[b * KCAND + e] = v;
    sidx[b * KCAND + e] = id;
    bool valid = v > 0.0f;
    float b0 = 0, b1 = 0, b2 = 0, b3 = 0; int lab = 0;
    if (valid) {
        int prior = id / NC1;
        lab = id % NC1 + 1;
        const float* src = boxes + ((size_t)b * P + prior) * 4;
        b0 = src[0]; b1 = src[1]; b2 = src[2]; b3 = src[3];
    }
    float* cb = cboxes + ((size_t)b * KCAND + e) * 4;
    cb[0] = b0; cb[1] = b1; cb[2] = b2; cb[3] = b3;
    clabels[b * KCAND + e] = lab;
    float m = valid ? fmaxf(fmaxf(b0, b1), fmaxf(b2, b3)) : 0.0f;
    red[tid] = fmaxf(m, 0.0f);
    __syncthreads();
    for (int s2 = 512; s2 > 0; s2 >>= 1) {
        if (tid < s2) red[tid] = fmaxf(red[tid], red[tid + s2]);
        __syncthreads();
    }
    float off = (red[0] + 1.0f) * (float)lab;
    float* ob = oboxes + ((size_t)b * KCAND + e) * 4;
    ob[0] = b0 + off; ob[1] = b1 + off; ob[2] = b2 + off; ob[3] = b3 + off;
}

// ---------------------------------------------------------------------------
// Kernel 8: 1024x1024 upper-triangular IoU suppression bitmask (cols > row).
// The 16 KB offset-box tile is staged in LDS once per block instead of
// re-reading it ~1024x from global per thread.
// ---------------------------------------------------------------------------
__global__ void k_mask(const float* __restrict__ oboxes,
                       unsigned* __restrict__ mask) {
    __shared__ float tile[KCAND * 4];
    int b = blockIdx.x >> 2;
    int r = (blockIdx.x & 3) * 256 + threadIdx.x;
    const float* ob = oboxes + (size_t)b * KCAND * 4;
    for (int i = threadIdx.x; i < KCAND * 4; i += 256) tile[i] = ob[i];
    __syncthreads();
    float x1 = tile[r * 4 + 0], y1 = tile[r * 4 + 1];
    float x2 = tile[r * 4 + 2], y2 = tile[r * 4 + 3];
    float ar = (x2 - x1) * (y2 - y1);
    unsigned* mrow = mask + ((size_t)b * KCAND + r) * 32;
    for (int w = 0; w < 32; ++w) {
        unsigned bits = 0u;
#pragma unroll
        for (int j2 = 0; j2 < 32; ++j2) {
            int j = w * 32 + j2;
            if (j > r) {
                float u1 = tile[j * 4 + 0], v1 = tile[j * 4 + 1];
                float u2 = tile[j * 4 + 2], v2 = tile[j * 4 + 3];
                float aj = (u2 - u1) * (v2 - v1);
                float ix = fmaxf(fminf(x2, u2) - fmaxf(x1, u1), 0.0f);
                float iy = fmaxf(fminf(y2, v2) - fmaxf(y1, v1), 0.0f);
                float inter = ix * iy;
                float iou = inter / (ar + aj - inter + 1e-9f);
                if (iou > NMS_THRESH) bits |= (1u << j2);
            }
        }
        mrow[w] = bits;
    }
}

// ---------------------------------------------------------------------------
// Kernel 9: greedy NMS, one wave32 per image. Lane l owns suppression word l;
// the "is row i suppressed?" bit is broadcast with __shfl each iteration --
// exact fori_loop semantics (ascending i == descending score).
// ---------------------------------------------------------------------------
__global__ void k_nms(const unsigned* __restrict__ mask,
                      const float* __restrict__ sval,
                      unsigned* __restrict__ keep) {
    int b = blockIdx.x, lane = threadIdx.x;
    unsigned remv = 0u, keepw = 0u;
    const unsigned* m  = mask + (size_t)b * KCAND * 32;
    const float*    sv = sval + (size_t)b * KCAND;
    for (int i = 0; i < KTOP; ++i) {
        unsigned rw = __shfl(remv, i >> 5, 32);
        bool removed = (rw >> (i & 31)) & 1u;
        bool kept = !removed && (sv[i] > 0.0f);   // wave-uniform
        if (kept) {
            remv |= m[(size_t)i * 32 + lane];
            if (lane == (i >> 5)) keepw |= (1u << (i & 31));
        }
    }
    keep[b * 32 + lane] = keepw;
}

// ---------------------------------------------------------------------------
// Kernel 10: output. Since candidates are sorted descending, top-100 of the
// kept scores == first 100 kept entries in order. Pad 0/0/-1.
// ---------------------------------------------------------------------------
__global__ void k_out(const float* __restrict__ sval,
                      const int* __restrict__ clabels,
                      const float* __restrict__ cboxes,
                      const unsigned* __restrict__ keep,
                      float* __restrict__ out) {
    int b = blockIdx.x;
    if (threadIdx.x != 0) return;
    float* obx = out + (size_t)b * DETS * 4;
    float* osc = out + (size_t)B * DETS * 4 + (size_t)b * DETS;
    float* olb = out + (size_t)B * DETS * 4 + (size_t)B * DETS + (size_t)b * DETS;
    int n = 0;
    for (int i = 0; i < KTOP && n < DETS; ++i) {
        if ((keep[b * 32 + (i >> 5)] >> (i & 31)) & 1u) {
            const float* cb = cboxes + ((size_t)b * KCAND + i) * 4;
            obx[n * 4 + 0] = cb[0]; obx[n * 4 + 1] = cb[1];
            obx[n * 4 + 2] = cb[2]; obx[n * 4 + 3] = cb[3];
            osc[n] = sval[b * KCAND + i];
            olb[n] = (float)clabels[b * KCAND + i];
            ++n;
        }
    }
    for (; n < DETS; ++n) {
        obx[n * 4 + 0] = 0.0f; obx[n * 4 + 1] = 0.0f;
        obx[n * 4 + 2] = 0.0f; obx[n * 4 + 3] = 0.0f;
        osc[n] = 0.0f; olb[n] = -1.0f;
    }
}

// ---------------------------------------------------------------------------
extern "C" void kernel_launch(void* const* d_in, const int* in_sizes, int n_in,
                              void* d_out, int out_size, void* d_ws, size_t ws_size,
                              hipStream_t stream) {
    (void)in_sizes; (void)n_in; (void)out_size; (void)ws_size;
    const float* logits = (const float*)d_in[0];
    const float* rel    = (const float*)d_in[1];
    const float* priors = (const float*)d_in[2];
    const int*   tsz    = (const int*)d_in[3];
    float* out = (float*)d_out;

    // workspace carve-up (256B aligned), total ~10.6 MB
    char* base = (char*)d_ws;
    size_t off = 0;
    auto alloc = [&](size_t bytes) -> char* {
        char* p = base + off;
        off = (off + bytes + 255) & ~(size_t)255;
        return p;
    };
    float*    boxes   = (float*)   alloc((size_t)B * P * 4 * 4);
    float*    maxlog  = (float*)   alloc((size_t)B * P * 4);
    float*    sumexp  = (float*)   alloc((size_t)B * P * 4);
    unsigned* hist    = (unsigned*)alloc((size_t)B * HBINS * 4);
    int*      cnt1    = (int*)     alloc((size_t)B * 4);
    int*      cnt2    = (int*)     alloc((size_t)B * 4);
    int*      tstar   = (int*)     alloc((size_t)B * 4);
    float*    l1v     = (float*)   alloc((size_t)B * KCAND * 4);
    int*      l1i     = (int*)     alloc((size_t)B * KCAND * 4);
    float*    l2v     = (float*)   alloc((size_t)B * KCAND * 4);
    int*      l2i     = (int*)     alloc((size_t)B * KCAND * 4);
    float*    sval    = (float*)   alloc((size_t)B * KCAND * 4);
    int*      sidx    = (int*)     alloc((size_t)B * KCAND * 4);
    float*    cboxes  = (float*)   alloc((size_t)B * KCAND * 4 * 4);
    int*      clabels = (int*)     alloc((size_t)B * KCAND * 4);
    float*    oboxes  = (float*)   alloc((size_t)B * KCAND * 4 * 4);
    unsigned* maskbuf = (unsigned*)alloc((size_t)B * KCAND * 32 * 4);
    unsigned* keepb   = (unsigned*)alloc((size_t)B * 32 * 4);
    (void)sidx;

    int bpThreads = B * P;
    int bpBlocks = (bpThreads + 255) / 256;

    k_decode<<<bpBlocks, 256, 0, stream>>>(rel, priors, tsz, boxes);

    int totalWaves = B * TILES_PER_IMG;               // 8736
    int smBlocks = (totalWaves + 7) / 8;              // 8 waves / 256-thread block
    k_softmax_wmma<<<smBlocks, 256, 0, stream>>>(logits, sumexp, maxlog);

    int zN = B * HBINS;
    k_zero<<<(zN + 255) / 256, 256, 0, stream>>>(hist, cnt1, cnt2);

    k_hist<<<bpBlocks, 256, 0, stream>>>(logits, sumexp, maxlog, boxes, hist);
    k_findt<<<B, 1024, 0, stream>>>(hist, tstar);
    k_collect<<<bpBlocks, 256, 0, stream>>>(logits, sumexp, maxlog, boxes, tstar,
                                            cnt1, cnt2, l1v, l1i, l2v, l2i);
    k_sort<<<B, 1024, 0, stream>>>(cnt1, cnt2, l1v, l1i, l2v, l2i, boxes,
                                   sval, sidx, cboxes, clabels, oboxes);
    k_mask<<<B * 4, 256, 0, stream>>>(oboxes, maskbuf);
    k_nms<<<B, 32, 0, stream>>>(maskbuf, sval, keepb);
    k_out<<<B, 32, 0, stream>>>(sval, clabels, cboxes, keepb, out);
}
